// GraphHallucinationRCNN_28071906246720
// MI455X (gfx1250) — matrix operand order repaired and checked by reading, not compiled, not run on hardware
//
#include <hip/hip_runtime.h>
#include <math.h>

#define B_     2
#define N_     384
#define C_     1024
#define HREP   256
#define GHID   256
#define HEADS  4
#define TOPK   8
#define KPAD   1056              // 1028 padded up to a multiple of 32
#define E_     (N_ * (TOPK + 1)) // 3456 edges incl. self loops
#define NEG_SLOPE 0.2f

typedef __attribute__((ext_vector_type(16))) _Float16 v16h;
typedef __attribute__((ext_vector_type(8)))  float    v8f;

// ---------------------------------------------------------------------------
// WMMA fragment loaders (wave32, v_wmma_f32_16x16x32_f16 contiguous k order chosen here, same for A and B)
// A 16x32 f16: lanes 0-15 row M=lane, V0-7 K=0..15 (2/vgpr);
//              lanes 16-31 same rows, K=16..31.
// B 32x16 f16: lanes 0-15 col N=lane, V0-7 K=0..15 (2/vgpr);
//              lanes 16-31 same cols, K=16..31.
// ---------------------------------------------------------------------------
__device__ inline v16h load_frag_a(const _Float16* __restrict__ row, int k0, int lane) {
    const int koff = (lane >> 4) * 16;
    union { v16h v; uint4 q[2]; } u;
    u.q[0] = *reinterpret_cast<const uint4*>(row + k0 + koff);
    u.q[1] = *reinterpret_cast<const uint4*>(row + k0 + koff + 8);
    return u.v;
}
__device__ inline v16h load_frag_b(const _Float16* __restrict__ col, int k0, int lane) {
    const int koff = (lane >> 4) * 16;
    union { v16h v; uint4 q[2]; } u;
    u.q[0] = *reinterpret_cast<const uint4*>(col + k0 + koff);
    u.q[1] = *reinterpret_cast<const uint4*>(col + k0 + koff + 8);
    return u.v;
}

__device__ inline float wave_sum(float v) {
    #pragma unroll
    for (int m = 16; m >= 1; m >>= 1) v += __shfl_xor(v, m, 32);
    return v;
}
__device__ inline float wave_max(float v) {
    #pragma unroll
    for (int m = 16; m >= 1; m >>= 1) v = fmaxf(v, __shfl_xor(v, m, 32));
    return v;
}
__device__ inline float lrelu(float x) { return x > 0.f ? x : NEG_SLOPE * x; }

// edge e -> (src, tgt).  e < N*TOPK: src=e/TOPK, tgt=nbr[e]; else self loop.
__device__ inline void edge_st(int e, const int* __restrict__ nb, int& src, int& tgt) {
    if (e < N_ * TOPK) { src = e / TOPK; tgt = nb[e]; }
    else               { src = e - N_ * TOPK; tgt = src; }
}

// ---------------------------------------------------------------------------
// K0: convert/transpose weights to f16 once.
//   Wa[n][k] = fc1_w[k][n]              (256 x 1024)
//   Wb[n][k] = fc1_w[1024+k][n]         (256 x 1024)
//   W1[n][k] = gat1_w[k][n], 0-padded   (1024 x KPAD)
// ---------------------------------------------------------------------------
__global__ void conv_weights(const float* __restrict__ fc1_w,
                             const float* __restrict__ gat1_w,
                             _Float16* __restrict__ Wa, _Float16* __restrict__ Wb,
                             _Float16* __restrict__ W1) {
    const int tid = blockIdx.x * blockDim.x + threadIdx.x;
    const int stride = gridDim.x * blockDim.x;
    for (int i = tid; i < HREP * C_; i += stride) {
        int n = i / C_, k = i % C_;
        Wa[i] = (_Float16)fc1_w[(long)k * HREP + n];
        Wb[i] = (_Float16)fc1_w[(long)(C_ + k) * HREP + n];
    }
    for (int i = tid; i < C_ * KPAD; i += stride) {
        int n = i / KPAD, k = i % KPAD;
        W1[i] = (k < C_ + 4) ? (_Float16)gat1_w[(long)k * C_ + n] : (_Float16)0.f;
    }
}

// ---------------------------------------------------------------------------
// K1: x16[b][n][0..1023] = feats, [1024..1027] = box geometry, [..1055] = 0
// ---------------------------------------------------------------------------
__global__ void build_x(const float* __restrict__ feats,
                        const float* __restrict__ boxes,
                        _Float16* __restrict__ x16) {
    const int n = blockIdx.x, b = blockIdx.y;
    const float* f  = feats + ((long)b * N_ + n) * C_;
    const float* bx = boxes + ((long)b * N_ + n) * 4;
    _Float16* xr = x16 + ((long)b * N_ + n) * KPAD;
    const float n0 = bx[0] / 800.f, n1 = bx[1] / 800.f;
    const float n2 = bx[2] / 800.f, n3 = bx[3] / 800.f;
    const float g[4] = { n0, n1, n2 - n0, n3 - n1 };
    for (int c = threadIdx.x; c < KPAD; c += blockDim.x) {
        float v = (c < C_) ? f[c] : (c < C_ + 4 ? g[c - C_] : 0.f);
        xr[c] = (_Float16)v;
    }
}

// ---------------------------------------------------------------------------
// K2: WMMA GEMM  C[M x N] = A16[M x K] * Bt16[N x K]^T   (f32 accum)
// one wave computes a 16(M) x 64(N) tile: one A fragment is reused against
// 4 B fragments -> 4 v_wmma per k-step, ~1.6x fewer bytes/FLOP than 16x16.
// grid = (N/64, M/16, batch).  N must be a multiple of 64.
// ---------------------------------------------------------------------------
__global__ __launch_bounds__(32)
void gemm_wmma_f16(const _Float16* __restrict__ A, int lda, long aStride,
                   const _Float16* __restrict__ Bt, int ldb,
                   float* __restrict__ Cm, int ldc, long cStride, int K) {
    const int lane = threadIdx.x & 31;
    const int n0 = blockIdx.x * 64;
    const int m0 = blockIdx.y * 16;
    const int b  = blockIdx.z;
    const _Float16* Arow = A + b * aStride + (long)(m0 + (lane & 15)) * lda;
    const _Float16* Bcol = Bt + (long)(n0 + (lane & 15)) * ldb;
    const long bstep = 16L * ldb;               // element offset: 16 rows of Bt (next 16 N cols)
    v8f acc0 = {}, acc1 = {}, acc2 = {}, acc3 = {};
    for (int k0 = 0; k0 < K; k0 += 32) {
        // speculative prefetch of the K+128 line (global_prefetch_b8);
        // drop-on-fault, so the tail overshoot is harmless.
        __builtin_prefetch(Arow + k0 + 128, 0, 0);
        __builtin_prefetch(Bcol + k0 + 128, 0, 0);
        v16h af = load_frag_a(Arow, k0, lane);
        v16h b0 = load_frag_b(Bcol,             k0, lane);
        v16h b1 = load_frag_b(Bcol + bstep,     k0, lane);
        v16h b2 = load_frag_b(Bcol + 2 * bstep, k0, lane);
        v16h b3 = load_frag_b(Bcol + 3 * bstep, k0, lane);
        acc0 = __builtin_amdgcn_wmma_f32_16x16x32_f16(
            false, af, false, b0, (short)0, acc0, false, false);
        acc1 = __builtin_amdgcn_wmma_f32_16x16x32_f16(
            false, af, false, b1, (short)0, acc1, false, false);
        acc2 = __builtin_amdgcn_wmma_f32_16x16x32_f16(
            false, af, false, b2, (short)0, acc2, false, false);
        acc3 = __builtin_amdgcn_wmma_f32_16x16x32_f16(
            false, af, false, b3, (short)0, acc3, false, false);
    }
    float* Cb = Cm + b * cStride;
    const int col   = lane & 15;
    const int mbase = (lane >> 4) * 8;
    #pragma unroll
    for (int v = 0; v < 8; ++v) {
        float* cr = Cb + (long)(m0 + mbase + v) * ldc + col;
        cr[n0 +  0] = acc0[v];
        cr[n0 + 16] = acc1[v];
        cr[n0 + 32] = acc2[v];
        cr[n0 + 48] = acc3[v];
    }
}

// ---------------------------------------------------------------------------
// K3: rel[i][j] = fc2_b + sum_k relu(p1[i,k]+p2[j,k]+|bx_i-bx_j|.Wg[:,k]+b[k])*w2[k]
// diag gets -1e6.  block = 256 (8 waves), wave w handles j = w, w+8, ...
// ---------------------------------------------------------------------------
__global__ __launch_bounds__(256)
void rel_kernel(const float* __restrict__ p1, const float* __restrict__ p2,
                const float* __restrict__ boxes, const float* __restrict__ fc1_w,
                const float* __restrict__ fc1_b, const float* __restrict__ fc2_w,
                const float* __restrict__ fc2_b, float* __restrict__ rel) {
    __shared__ float sp1[HREP], sb[HREP], sw2[HREP], sWg[4][HREP];
    const int i = blockIdx.x, b = blockIdx.y;
    const float* p1r = p1 + ((long)b * N_ + i) * HREP;
    for (int k = threadIdx.x; k < HREP; k += 256) {
        sp1[k] = p1r[k]; sb[k] = fc1_b[k]; sw2[k] = fc2_w[k];
        #pragma unroll
        for (int c = 0; c < 4; ++c)
            sWg[c][k] = fc1_w[(long)(2 * C_ + c) * HREP + k];
    }
    __syncthreads();
    const float* bxb = boxes + (long)b * N_ * 4;
    const float bi0 = bxb[i * 4 + 0], bi1 = bxb[i * 4 + 1];
    const float bi2 = bxb[i * 4 + 2], bi3 = bxb[i * 4 + 3];
    const int wave = threadIdx.x >> 5, lane = threadIdx.x & 31;
    const float c2b = fc2_b[0];
    float* relr = rel + ((long)b * N_ + i) * N_;
    for (int j = wave; j < N_; j += 8) {
        const float d0 = fabsf(bi0 - bxb[j * 4 + 0]);
        const float d1 = fabsf(bi1 - bxb[j * 4 + 1]);
        const float d2 = fabsf(bi2 - bxb[j * 4 + 2]);
        const float d3 = fabsf(bi3 - bxb[j * 4 + 3]);
        const float* p2r = p2 + ((long)b * N_ + j) * HREP;
        float acc = 0.f;
        for (int k = lane; k < HREP; k += 32) {
            float v = sp1[k] + p2r[k] + d0 * sWg[0][k] + d1 * sWg[1][k]
                    + d2 * sWg[2][k] + d3 * sWg[3][k] + sb[k];
            acc += fmaxf(v, 0.f) * sw2[k];
        }
        acc = wave_sum(acc);
        if (lane == 0)
            relr[j] = acc + c2b - (i == j ? 1.0e6f : 0.f);
    }
}

// ---------------------------------------------------------------------------
// K4: top-8 per row (ties -> smallest index, matching jax top_k order)
// ---------------------------------------------------------------------------
__global__ __launch_bounds__(32)
void topk_kernel(const float* __restrict__ rel, int* __restrict__ nbr) {
    const int i = blockIdx.x, b = blockIdx.y, lane = threadIdx.x;
    const float* r = rel + ((long)b * N_ + i) * N_;
    float v[12];
    const int base = lane * 12;                 // 384 / 32 = 12 per lane
    #pragma unroll
    for (int t = 0; t < 12; ++t) v[t] = r[base + t];
    int* out = nbr + ((long)b * N_ + i) * TOPK;
    for (int k = 0; k < TOPK; ++k) {
        float bestv = -INFINITY; int besti = 0x7fffffff;
        #pragma unroll
        for (int t = 0; t < 12; ++t)
            if (v[t] > bestv || (v[t] == bestv && base + t < besti)) {
                bestv = v[t]; besti = base + t;
            }
        #pragma unroll
        for (int m = 16; m >= 1; m >>= 1) {
            float ov = __shfl_xor(bestv, m, 32);
            int   oi = __shfl_xor(besti, m, 32);
            if (ov > bestv || (ov == bestv && oi < besti)) { bestv = ov; besti = oi; }
        }
        if (lane == 0) out[k] = besti;
        if (besti >= base && besti < base + 12) v[besti - base] = -INFINITY;
    }
}

// ---------------------------------------------------------------------------
// K5: GAT1 attention coefficients  a_s[n,h], a_d[n,h]   (4 waves = 4 heads)
// ---------------------------------------------------------------------------
__global__ __launch_bounds__(128)
void att1_kernel(const float* __restrict__ xw, const float* __restrict__ att_src,
                 const float* __restrict__ att_dst,
                 float* __restrict__ a_s, float* __restrict__ a_d) {
    const int n = blockIdx.x, b = blockIdx.y;
    const int h = threadIdx.x >> 5, lane = threadIdx.x & 31;
    const float* x = xw + ((long)b * N_ + n) * (HEADS * GHID) + h * GHID;
    float ss = 0.f, sd = 0.f;
    for (int c = lane; c < GHID; c += 32) {
        const float xv = x[c];
        ss += xv * att_src[h * GHID + c];
        sd += xv * att_dst[h * GHID + c];
    }
    ss = wave_sum(ss); sd = wave_sum(sd);
    if (lane == 0) {
        a_s[((long)b * N_ + n) * HEADS + h] = ss;
        a_d[((long)b * N_ + n) * HEADS + h] = sd;
    }
}

// ---------------------------------------------------------------------------
// K6/K9: per-target segment max + exp-sum over incoming edges (heads <= 4)
// one wave per target node; brute-force scan of the 3456 edges.
// ---------------------------------------------------------------------------
__global__ __launch_bounds__(32)
void edge_soft(const int* __restrict__ nbr, const float* __restrict__ a_s,
               const float* __restrict__ a_d, float* __restrict__ emax,
               float* __restrict__ edenom, int heads) {
    const int t = blockIdx.x, b = blockIdx.y, lane = threadIdx.x;
    const int* nb = nbr + (long)b * N_ * TOPK;
    const float* as = a_s + (long)b * N_ * heads;
    const float* ad = a_d + (long)b * N_ * heads;
    float mx[4] = { -INFINITY, -INFINITY, -INFINITY, -INFINITY };
    for (int e = lane; e < E_; e += 32) {
        int src, tgt; edge_st(e, nb, src, tgt);
        if (tgt != t) continue;
        for (int h = 0; h < heads; ++h)
            mx[h] = fmaxf(mx[h], lrelu(as[src * heads + h] + ad[t * heads + h]));
    }
    for (int h = 0; h < heads; ++h) mx[h] = wave_max(mx[h]);
    float dn[4] = { 0.f, 0.f, 0.f, 0.f };
    for (int e = lane; e < E_; e += 32) {
        int src, tgt; edge_st(e, nb, src, tgt);
        if (tgt != t) continue;
        for (int h = 0; h < heads; ++h) {
            const float ev = lrelu(as[src * heads + h] + ad[t * heads + h]);
            dn[h] += __expf(ev - mx[h]);
        }
    }
    for (int h = 0; h < heads; ++h) dn[h] = wave_sum(dn[h]);
    if (lane == 0)
        for (int h = 0; h < heads; ++h) {
            emax  [((long)b * N_ + t) * heads + h] = mx[h];
            edenom[((long)b * N_ + t) * heads + h] = dn[h];
        }
}

// ---------------------------------------------------------------------------
// K7: GAT1 aggregation: deterministic compaction of incoming edges into LDS
// (chunked scan + serial prefix), then per-channel weighted sum, bias, relu.
// ---------------------------------------------------------------------------
#define MAXDEG 400
__global__ __launch_bounds__(256)
void agg1_kernel(const int* __restrict__ nbr, const float* __restrict__ a_s,
                 const float* __restrict__ a_d, const float* __restrict__ emax,
                 const float* __restrict__ edenom, const float* __restrict__ xw,
                 const float* __restrict__ bias, float* __restrict__ h1) {
    __shared__ int   scnt[257];
    __shared__ int   ssrc[MAXDEG];
    __shared__ float salpha[MAXDEG * HEADS];
    const int t = blockIdx.x, b = blockIdx.y, tid = threadIdx.x;
    const int* nb = nbr + (long)b * N_ * TOPK;
    const int CH = (E_ + 255) / 256;
    const int e0 = tid * CH;
    const int e1 = min(e0 + CH, E_);
    int cnt = 0;
    for (int e = e0; e < e1; ++e) {
        int src, tgt; edge_st(e, nb, src, tgt);
        if (tgt == t) ++cnt;
    }
    scnt[tid] = cnt;
    __syncthreads();
    if (tid == 0) {
        int acc = 0;
        for (int i = 0; i < 256; ++i) { int c = scnt[i]; scnt[i] = acc; acc += c; }
        scnt[256] = acc;
    }
    __syncthreads();
    const float* as = a_s + (long)b * N_ * HEADS;
    const float* ad = a_d + (long)b * N_ * HEADS;
    const float* mx = emax   + ((long)b * N_ + t) * HEADS;
    const float* dn = edenom + ((long)b * N_ + t) * HEADS;
    int pos = scnt[tid];
    for (int e = e0; e < e1; ++e) {
        int src, tgt; edge_st(e, nb, src, tgt);
        if (tgt != t) continue;
        ssrc[pos] = src;
        #pragma unroll
        for (int h = 0; h < HEADS; ++h) {
            const float ev = lrelu(as[src * HEADS + h] + ad[t * HEADS + h]);
            salpha[pos * HEADS + h] = __expf(ev - mx[h]) / fmaxf(dn[h], 1e-16f);
        }
        ++pos;
    }
    __syncthreads();
    const int ne = scnt[256];
    const float* xwb = xw + (long)b * N_ * (HEADS * GHID);
    float* outr = h1 + ((long)b * N_ + t) * (HEADS * GHID);
    for (int ch = tid; ch < HEADS * GHID; ch += 256) {
        const int h = ch >> 8;            // / GHID
        float acc = 0.f;
        for (int p = 0; p < ne; ++p)
            acc += salpha[p * HEADS + h] * xwb[(long)ssrc[p] * (HEADS * GHID) + ch];
        outr[ch] = fmaxf(acc + bias[ch], 0.f);   // relu(h1)
    }
}

// ---------------------------------------------------------------------------
// K8: GAT2 projection (1024 -> 2) + attention coefficients (heads=1)
// ---------------------------------------------------------------------------
__global__ __launch_bounds__(32)
void gat2_prep(const float* __restrict__ h1, const float* __restrict__ w2,
               const float* __restrict__ att_s, const float* __restrict__ att_d,
               float* __restrict__ xw2, float* __restrict__ as2, float* __restrict__ ad2) {
    const int n = blockIdx.x, b = blockIdx.y, lane = threadIdx.x;
    const float* hr = h1 + ((long)b * N_ + n) * (HEADS * GHID);
    float s0 = 0.f, s1 = 0.f;
    for (int k = lane; k < HEADS * GHID; k += 32) {
        const float hv = hr[k];
        s0 += hv * w2[k * 2 + 0];
        s1 += hv * w2[k * 2 + 1];
    }
    s0 = wave_sum(s0); s1 = wave_sum(s1);
    if (lane == 0) {
        const long idx = (long)b * N_ + n;
        xw2[idx * 2 + 0] = s0; xw2[idx * 2 + 1] = s1;
        as2[idx] = s0 * att_s[0] + s1 * att_s[1];
        ad2[idx] = s0 * att_d[0] + s1 * att_d[1];
    }
}

// ---------------------------------------------------------------------------
// K10: GAT2 aggregation -> logits (B, N, 2)
// ---------------------------------------------------------------------------
__global__ __launch_bounds__(32)
void agg2_kernel(const int* __restrict__ nbr, const float* __restrict__ as2,
                 const float* __restrict__ ad2, const float* __restrict__ emax2,
                 const float* __restrict__ eden2, const float* __restrict__ xw2,
                 const float* __restrict__ b2, float* __restrict__ out) {
    const int t = blockIdx.x, b = blockIdx.y, lane = threadIdx.x;
    const int* nb = nbr + (long)b * N_ * TOPK;
    const float mx = emax2[(long)b * N_ + t];
    const float dn = fmaxf(eden2[(long)b * N_ + t], 1e-16f);
    float o0 = 0.f, o1 = 0.f;
    for (int e = lane; e < E_; e += 32) {
        int src, tgt; edge_st(e, nb, src, tgt);
        if (tgt != t) continue;
        const float ev = lrelu(as2[(long)b * N_ + src] + ad2[(long)b * N_ + t]);
        const float al = __expf(ev - mx) / dn;
        o0 += al * xw2[((long)b * N_ + src) * 2 + 0];
        o1 += al * xw2[((long)b * N_ + src) * 2 + 1];
    }
    o0 = wave_sum(o0); o1 = wave_sum(o1);
    if (lane == 0) {
        out[((long)b * N_ + t) * 2 + 0] = o0 + b2[0];
        out[((long)b * N_ + t) * 2 + 1] = o1 + b2[1];
    }
}

// ---------------------------------------------------------------------------
extern "C" void kernel_launch(void* const* d_in, const int* in_sizes, int n_in,
                              void* d_out, int out_size, void* d_ws, size_t ws_size,
                              hipStream_t stream) {
    const float* feats  = (const float*)d_in[0];
    const float* boxes  = (const float*)d_in[1];
    const float* fc1_w  = (const float*)d_in[2];
    const float* fc1_b  = (const float*)d_in[3];
    const float* fc2_w  = (const float*)d_in[4];
    const float* fc2_b  = (const float*)d_in[5];
    const float* gat1_w = (const float*)d_in[6];
    const float* g1_as  = (const float*)d_in[7];
    const float* g1_ad  = (const float*)d_in[8];
    const float* gat1_b = (const float*)d_in[9];
    const float* gat2_w = (const float*)d_in[10];
    const float* g2_as  = (const float*)d_in[11];
    const float* g2_ad  = (const float*)d_in[12];
    const float* gat2_b = (const float*)d_in[13];
    float* out = (float*)d_out;

    char* ws = (char*)d_ws;
    size_t off = 0;
    auto alloc = [&](size_t bytes) -> char* {
        char* p = ws + off;
        off = (off + bytes + 255) & ~(size_t)255;
        return p;
    };
    _Float16* Wa   = (_Float16*)alloc((size_t)HREP * C_ * 2);
    _Float16* Wb   = (_Float16*)alloc((size_t)HREP * C_ * 2);
    _Float16* W1   = (_Float16*)alloc((size_t)C_ * KPAD * 2);
    _Float16* x16  = (_Float16*)alloc((size_t)B_ * N_ * KPAD * 2);
    float* p1      = (float*)alloc((size_t)B_ * N_ * HREP * 4);
    float* p2      = (float*)alloc((size_t)B_ * N_ * HREP * 4);
    float* rel     = (float*)alloc((size_t)B_ * N_ * N_ * 4);
    int*   nbr     = (int*)  alloc((size_t)B_ * N_ * TOPK * 4);
    float* xw1     = (float*)alloc((size_t)B_ * N_ * HEADS * GHID * 4);
    float* a_s     = (float*)alloc((size_t)B_ * N_ * HEADS * 4);
    float* a_d     = (float*)alloc((size_t)B_ * N_ * HEADS * 4);
    float* emax1   = (float*)alloc((size_t)B_ * N_ * HEADS * 4);
    float* eden1   = (float*)alloc((size_t)B_ * N_ * HEADS * 4);
    float* h1      = (float*)alloc((size_t)B_ * N_ * HEADS * GHID * 4);
    float* xw2     = (float*)alloc((size_t)B_ * N_ * 2 * 4);
    float* as2     = (float*)alloc((size_t)B_ * N_ * 4);
    float* ad2     = (float*)alloc((size_t)B_ * N_ * 4);
    float* emax2   = (float*)alloc((size_t)B_ * N_ * 4);
    float* eden2   = (float*)alloc((size_t)B_ * N_ * 4);
    (void)ws_size; (void)in_sizes; (void)n_in; (void)out_size;

    // K0: weight conversion / transpose to f16
    conv_weights<<<2048, 256, 0, stream>>>(fc1_w, gat1_w, Wa, Wb, W1);
    // K1: build padded f16 node features
    build_x<<<dim3(N_, B_), 256, 0, stream>>>(feats, boxes, x16);
    // K2a/K2b: p1 = f@Wa, p2 = f@Wb  (K = 1024, first 1024 cols of x16)
    gemm_wmma_f16<<<dim3(HREP / 64, N_ / 16, B_), 32, 0, stream>>>(
        x16, KPAD, (long)N_ * KPAD, Wa, C_, p1, HREP, (long)N_ * HREP, C_);
    gemm_wmma_f16<<<dim3(HREP / 64, N_ / 16, B_), 32, 0, stream>>>(
        x16, KPAD, (long)N_ * KPAD, Wb, C_, p2, HREP, (long)N_ * HREP, C_);
    // K3: pairwise relation scores
    rel_kernel<<<dim3(N_, B_), 256, 0, stream>>>(
        p1, p2, boxes, fc1_w, fc1_b, fc2_w, fc2_b, rel);
    // K4: top-8 neighbors per row
    topk_kernel<<<dim3(N_, B_), 32, 0, stream>>>(rel, nbr);
    // K2c: xw1 = x @ gat1_w  (K = KPAD, zero-padded rows)
    gemm_wmma_f16<<<dim3((HEADS * GHID) / 64, N_ / 16, B_), 32, 0, stream>>>(
        x16, KPAD, (long)N_ * KPAD, W1, KPAD, xw1, HEADS * GHID,
        (long)N_ * HEADS * GHID, KPAD);
    // K5: attention coefficients
    att1_kernel<<<dim3(N_, B_), 128, 0, stream>>>(xw1, g1_as, g1_ad, a_s, a_d);
    // K6: segment softmax stats (4 heads)
    edge_soft<<<dim3(N_, B_), 32, 0, stream>>>(nbr, a_s, a_d, emax1, eden1, HEADS);
    // K7: GAT1 aggregate + bias + relu
    agg1_kernel<<<dim3(N_, B_), 256, 0, stream>>>(
        nbr, a_s, a_d, emax1, eden1, xw1, gat1_b, h1);
    // K8: GAT2 projection + attention coefficients
    gat2_prep<<<dim3(N_, B_), 32, 0, stream>>>(h1, gat2_w, g2_as, g2_ad, xw2, as2, ad2);
    // K9: segment softmax stats (1 head)
    edge_soft<<<dim3(N_, B_), 32, 0, stream>>>(nbr, as2, ad2, emax2, eden2, 1);
    // K10: final aggregation -> logits
    agg2_kernel<<<dim3(N_, B_), 32, 0, stream>>>(
        nbr, as2, ad2, emax2, eden2, xw2, gat2_b, out);
}
